// DenseDilatedKnnGraphDGL_83777632076276
// MI455X (gfx1250) — matrix operand order repaired and checked by reading, not compiled, not run on hardware
//
#include <hip/hip_runtime.h>
#include <hip/hip_bf16.h>

#define BB 64
#define CC 256
#define NN 1024
#define MAXD 3

typedef __attribute__((ext_vector_type(16))) __bf16 v16bf;
typedef __attribute__((ext_vector_type(8)))  float  v8f;

union Frag { v16bf v; uint4 q[2]; };

// CDNA5 async memory->LDS copy (ASYNCcnt-tracked, no VGPR round trip).
// vdst = 32-bit LDS byte address (low 32 bits of the generic pointer per the
// LDS aperture rule), vaddr = 64-bit global address.
#define ASYNC_LDS_B128(ldsptr, gptr)                                        \
    asm volatile("global_load_async_to_lds_b128 %0, %1, off"                \
                 :: "v"((unsigned)(uintptr_t)(ldsptr)), "v"(gptr) : "memory")

// ---------------------------------------------------------------------------
// prep: x (B,C,N) f32  ->  pts (B,N,C) bf16 (row-major per point) + sq (B,N)
// Coalesced reads (consecutive n), LDS transpose, coalesced 16B stores.
// ---------------------------------------------------------------------------
__global__ void __launch_bounds__(256)
prep_kernel(const float* __restrict__ x, __bf16* __restrict__ pts,
            float* __restrict__ sq) {
    const int b  = blockIdx.x >> 5;          // N/32 = 32 tiles per batch
    const int n0 = (blockIdx.x & 31) << 5;
    const int t  = threadIdx.x;
    const int nl = t & 31;                   // local point
    const int cg = t >> 5;                   // channel group 0..7

    __shared__ __bf16 tile[32][CC + 8];      // +8 bf16 pad: kills bank conflicts
    __shared__ float  sqp[8][32];

    const float* xb = x + (size_t)b * CC * NN;
    float acc = 0.f;
    for (int c8 = 0; c8 < CC; c8 += 8) {
        int   c = c8 + cg;
        float v = xb[(size_t)c * NN + (n0 + nl)];   // coalesced in nl
        acc += v * v;
        tile[nl][c] = (__bf16)v;
    }
    sqp[cg][nl] = acc;
    __syncthreads();

    if (t < 32) {
        float s = 0.f;
        for (int i = 0; i < 8; ++i) s += sqp[i][t];
        sq[(size_t)b * NN + n0 + t] = s;
    }

    // 32 rows x 512B = 1024 chunks of 16B, 4 per thread, coalesced stores
    for (int chunk = t; chunk < 32 * (CC / 8); chunk += 256) {
        int row = chunk >> 5;
        int off = chunk & 31;
        uint4 val = *(const uint4*)&tile[row][off * 8];
        *(uint4*)(pts + ((size_t)(b * NN + n0 + row)) * CC + off * 8) = val;
    }
}

// ---------------------------------------------------------------------------
// knn: one block = (batch b, 16-row tile n0). 8 waves; wave w owns column
// tiles [8w .. 8w+7]. bf16 WMMA Gram tiles -> d2 panel in LDS -> per-row
// sorted top-(9*dil) extraction, keep every dil-th neighbor.
// ---------------------------------------------------------------------------
__global__ void __launch_bounds__(256)
knn_kernel(const __bf16* __restrict__ pts, const float* __restrict__ sq,
           const int* __restrict__ layer,
           int* __restrict__ out_src, int* __restrict__ out_dst) {
    const int b    = blockIdx.x >> 6;        // N/16 = 64 row tiles per batch
    const int n0   = (blockIdx.x & 63) << 4;
    const int t    = threadIdx.x;
    const int lane = t & 31;
    const int wave = t >> 5;

    __shared__ __bf16 arow[16][CC + 8];      // A tile rows (padded)
    __shared__ float  d2s[16][NN];           // 64KB distance panel
    __shared__ float  sqa[NN];

    const __bf16* Pb = pts + (size_t)b * NN * CC;

    // stage A rows (16 x 512B = 512 chunks) + sqa (256 chunks) via the CDNA5
    // async-to-LDS path; one s_wait_asynccnt per wave, then block barrier.
    for (int chunk = t; chunk < 16 * (CC / 8); chunk += 256) {
        int row = chunk >> 5, off = chunk & 31;
        ASYNC_LDS_B128(&arow[row][off * 8],
                       Pb + (size_t)(n0 + row) * CC + off * 8);
    }
    ASYNC_LDS_B128(&sqa[t * 4], sq + (size_t)b * NN + t * 4);
    asm volatile("s_wait_asynccnt 0" ::: "memory");
    __syncthreads();

    const int hi = lane >> 4;                // lane half
    const int nL = lane & 15;
    float sA[8];
    for (int r = 0; r < 8; ++r) sA[r] = sqa[n0 + r + 8 * hi];

    for (int mt = wave * 8; mt < wave * 8 + 8; ++mt) {
        const int m0 = mt << 4;
        // B fragment: lane = column point m0+nL; K run split by lane half
        const __bf16* Brow = Pb + (size_t)(m0 + nL) * CC + (hi << 3);
        __builtin_prefetch(Brow + 16 * CC, 0, 3);   // next column tile, WGP scope
        v8f acc = {};
        for (int k0 = 0; k0 < CC; k0 += 32) {
            Frag a, bf;
            const int ak = k0 + (hi << 3);   // A: lo half K{0..7,16..23}, hi {8..15,24..31}
            a.q[0]  = *(const uint4*)&arow[nL][ak];
            a.q[1]  = *(const uint4*)&arow[nL][ak + 16];
            bf.q[0] = *(const uint4*)(Brow + k0);
            bf.q[1] = *(const uint4*)(Brow + k0 + 16);
            acc = __builtin_amdgcn_wmma_f32_16x16x32_bf16(
                false, a.v, false, bf.v, (short)0, acc, false, false);
        }
        const float sB = sqa[m0 + nL];
        for (int r = 0; r < 8; ++r)          // C/D layout: vgpr r -> M = r + 8*hi, N = nL
            d2s[r + 8 * hi][m0 + nL] = sA[r] + sB - 2.0f * acc[r];
    }
    __syncthreads();

    int dil = layer[0] / 4 + 1;
    if (dil > MAXD) dil = MAXD;
    const int kd = 9 * dil;
    const float INF = __builtin_inff();

    // wave w handles rows 2w, 2w+1
    for (int rr = wave * 2; rr < wave * 2 + 2; ++rr) {
        float* d2row = &d2s[rr][0];
        const int gdst = b * NN + n0 + rr;

        // each lane caches the min over its strided 32-element slice
        float lv = INF; int li = lane;
        for (int i = 0; i < NN / 32; ++i) {
            int idx = lane + (i << 5);
            float v = d2row[idx];
            if (v < lv) { lv = v; li = idx; }   // strict <: lowest-index tie-break
        }
        for (int k = 0; k < kd; ++k) {
            float bv = lv; int bi = li;
            for (int off = 16; off > 0; off >>= 1) {
                float ov = __shfl_xor(bv, off, 32);
                int   oi = __shfl_xor(bi, off, 32);
                if (ov < bv || (ov == bv && oi < bi)) { bv = ov; bi = oi; }
            }
            if ((k % dil) == 0 && lane == 0) {
                int j = k / dil;
                out_src[(size_t)gdst * 9 + j] = b * NN + bi;
                out_dst[(size_t)gdst * 9 + j] = gdst;
            }
            if (li == bi) {                  // winner lane: invalidate + rescan
                d2row[bi] = INF;
                lv = INF; li = lane;
                for (int i = 0; i < NN / 32; ++i) {
                    int idx = lane + (i << 5);
                    float v = d2row[idx];
                    if (v < lv) { lv = v; li = idx; }
                }
            }
        }
    }
}

extern "C" void kernel_launch(void* const* d_in, const int* in_sizes, int n_in,
                              void* d_out, int out_size, void* d_ws, size_t ws_size,
                              hipStream_t stream) {
    const float* x     = (const float*)d_in[0];
    const int*   layer = (const int*)d_in[1];

    __bf16* pts  = (__bf16*)d_ws;                                    // 32 MB
    float*  sqb  = (float*)((char*)d_ws + (size_t)BB * NN * CC * 2); // 256 KB
    int*    out  = (int*)d_out;
    int*    osrc = out;
    int*    odst = out + (size_t)BB * NN * 9;

    prep_kernel<<<BB * (NN / 32), 256, 0, stream>>>(x, pts, sqb);
    knn_kernel <<<BB * (NN / 16), 256, 0, stream>>>(pts, sqb, layer, osrc, odst);
}
